// Net0_72791105732853
// MI455X (gfx1250) — matrix operand order repaired and checked by reading, not compiled, not run on hardware
//
#include <hip/hip_runtime.h>
#include <hip/hip_bf16.h>

#define N_NODES 100000
#define N_EDGES 1600000
#define IN_F    50
#define HID     64
#define NCLS    41
#define KPAD    64   // K padded (50 -> 64 for layer1; layer2 K=64 exact)
#define CPAD    48   // class dim padded 41 -> 48 (3 col tiles)

typedef __bf16 bf16;
typedef __attribute__((ext_vector_type(16))) __bf16 v16bf;
typedef __attribute__((ext_vector_type(8)))  __bf16 v8bf;
typedef __attribute__((ext_vector_type(8)))  float  v8f;

// ---------- helpers ----------

__device__ __forceinline__ void atomic_add_f32(float* p, float v) {
    __hip_atomic_fetch_add(p, v, __ATOMIC_RELAXED, __HIP_MEMORY_SCOPE_AGENT);
}

// A fragment (16x32 bf16, wave32): lane m=L%16, hi=L/16 -> one contiguous
// 16-element K run at [kc*32 + hi*16] of row m.
__device__ __forceinline__ v16bf load_a_frag(const bf16* __restrict__ row, int kbase) {
    union { v16bf v; v8bf h[2]; } u;
    u.h[0] = *(const v8bf*)(row + kbase);
    u.h[1] = *(const v8bf*)(row + kbase + 8);
    return u.v;
}

// ---------- kernels ----------

// Pack weights column-major bf16 with zero padding:
// Wp1{l,r}: [64 cols][64 k], Wp2{l,r}: [48 cols][64 k]
__global__ void k_pack_weights(const float* __restrict__ W1l, const float* __restrict__ W1r,
                               const float* __restrict__ W2l, const float* __restrict__ W2r,
                               bf16* __restrict__ Wp1l, bf16* __restrict__ Wp1r,
                               bf16* __restrict__ Wp2l, bf16* __restrict__ Wp2r) {
    int i = blockIdx.x * blockDim.x + threadIdx.x;
    const int s1 = HID * KPAD;   // 4096
    const int s2 = CPAD * KPAD;  // 3072
    if (i < s1) {
        int n = i >> 6, k = i & 63;
        Wp1l[i] = (k < IN_F) ? (bf16)W1l[k * HID + n] : (bf16)0.0f;
    } else if (i < 2 * s1) {
        int j = i - s1; int n = j >> 6, k = j & 63;
        Wp1r[j] = (k < IN_F) ? (bf16)W1r[k * HID + n] : (bf16)0.0f;
    } else if (i < 2 * s1 + s2) {
        int j = i - 2 * s1; int n = j >> 6, k = j & 63;
        Wp2l[j] = (n < NCLS) ? (bf16)W2l[k * NCLS + n] : (bf16)0.0f;
    } else if (i < 2 * s1 + 2 * s2) {
        int j = i - 2 * s1 - s2; int n = j >> 6, k = j & 63;
        Wp2r[j] = (n < NCLS) ? (bf16)W2r[k * NCLS + n] : (bf16)0.0f;
    }
}

// x (f32 [N][50]) -> xb (bf16 [N][64], zero padded); zero agg (f32 [N][64])
__global__ void k_prep(const float* __restrict__ x, bf16* __restrict__ xb,
                       float* __restrict__ agg) {
    int i = blockIdx.x * blockDim.x + threadIdx.x;
    if (i >= N_NODES * KPAD) return;
    int node = i >> 6, f = i & 63;
    xb[i]  = (f < IN_F) ? (bf16)x[node * IN_F + f] : (bf16)0.0f;
    agg[i] = 0.0f;
}

// edge scatter layer 1: agg[dst] += x[src] (f32 atomics)
__global__ void k_scatter1(const float* __restrict__ x, const int* __restrict__ src,
                           const int* __restrict__ dst, float* __restrict__ agg) {
    int i = blockIdx.x * blockDim.x + threadIdx.x;
    if (i >= N_EDGES * 64) return;
    int e = i >> 6, f = i & 63;
    if (f >= IN_F) return;
    long s = src[e], d = dst[e];
    atomic_add_f32(&agg[d * KPAD + f], x[s * IN_F + f]);
}

// agg (f32) -> aggb (bf16), and re-zero agg for the next aggregation pass
__global__ void k_cvt_zero(float* __restrict__ agg, bf16* __restrict__ ab) {
    int i = blockIdx.x * blockDim.x + threadIdx.x;
    if (i >= N_NODES * KPAD) return;
    ab[i]  = (bf16)agg[i];
    agg[i] = 0.0f;
}

// GEMM1: hb = relu(aggb @ W1l + xb @ W1r + b1), output bf16 [N][64]
// grid = 6250 row tiles, block = 128 (4 waves, one 16-col tile each)
__global__ void __launch_bounds__(128)
k_gemm1(const bf16* __restrict__ aggb, const bf16* __restrict__ xb,
        const bf16* __restrict__ Wp1l, const bf16* __restrict__ Wp1r,
        const float* __restrict__ b1, bf16* __restrict__ hb) {
    int lane = threadIdx.x & 31;
    int wt   = threadIdx.x >> 5;      // column tile 0..3
    int row0 = blockIdx.x * 16;
    int m    = lane & 15;
    int hi   = lane >> 4;
    int ncol = wt * 16 + (lane & 15);

    const bf16* arow = aggb + (size_t)(row0 + m) * KPAD;
    const bf16* xrow = xb   + (size_t)(row0 + m) * KPAD;
    const bf16* wl   = Wp1l + (size_t)ncol * KPAD;
    const bf16* wr   = Wp1r + (size_t)ncol * KPAD;

    v8f acc = {};
#pragma unroll
    for (int kc = 0; kc < 2; ++kc) {
        v16bf a = load_a_frag(arow, kc * 32 + hi * 16);
        v16bf b = *(const v16bf*)(wl + kc * 32 + hi * 16);
        acc = __builtin_amdgcn_wmma_f32_16x16x32_bf16(false, a, false, b, (short)0, acc, false, false);
    }
#pragma unroll
    for (int kc = 0; kc < 2; ++kc) {
        v16bf a = load_a_frag(xrow, kc * 32 + hi * 16);
        v16bf b = *(const v16bf*)(wr + kc * 32 + hi * 16);
        acc = __builtin_amdgcn_wmma_f32_16x16x32_bf16(false, a, false, b, (short)0, acc, false, false);
    }
    float bias = b1[ncol];
#pragma unroll
    for (int i = 0; i < 8; ++i) {
        float v = acc[i] + bias;
        v = v > 0.0f ? v : 0.0f;               // ReLU fused
        hb[(size_t)(row0 + hi * 8 + i) * HID + ncol] = (bf16)v;
    }
}

// edge scatter layer 2: agg[dst] += (float)hb[src]
__global__ void k_scatter2(const bf16* __restrict__ hb, const int* __restrict__ src,
                           const int* __restrict__ dst, float* __restrict__ agg) {
    int i = blockIdx.x * blockDim.x + threadIdx.x;
    if (i >= N_EDGES * 64) return;
    int e = i >> 6, f = i & 63;
    long s = src[e], d = dst[e];
    atomic_add_f32(&agg[d * KPAD + f], (float)hb[s * HID + f]);
}

// GEMM2: logits = agg2b @ W2l + hb @ W2r + b2, output f32 [N][48]
// grid = 6250 row tiles, block = 96 (3 waves, one 16-col tile each)
__global__ void __launch_bounds__(96)
k_gemm2(const bf16* __restrict__ aggb, const bf16* __restrict__ hb,
        const bf16* __restrict__ Wp2l, const bf16* __restrict__ Wp2r,
        const float* __restrict__ b2, float* __restrict__ logits) {
    int lane = threadIdx.x & 31;
    int wt   = threadIdx.x >> 5;      // column tile 0..2
    int row0 = blockIdx.x * 16;
    int m    = lane & 15;
    int hi   = lane >> 4;
    int ncol = wt * 16 + (lane & 15);

    const bf16* arow = aggb + (size_t)(row0 + m) * KPAD;
    const bf16* hrow = hb   + (size_t)(row0 + m) * HID;
    const bf16* wl   = Wp2l + (size_t)ncol * KPAD;
    const bf16* wr   = Wp2r + (size_t)ncol * KPAD;

    v8f acc = {};
#pragma unroll
    for (int kc = 0; kc < 2; ++kc) {
        v16bf a = load_a_frag(arow, kc * 32 + hi * 16);
        v16bf b = *(const v16bf*)(wl + kc * 32 + hi * 16);
        acc = __builtin_amdgcn_wmma_f32_16x16x32_bf16(false, a, false, b, (short)0, acc, false, false);
    }
#pragma unroll
    for (int kc = 0; kc < 2; ++kc) {
        v16bf a = load_a_frag(hrow, kc * 32 + hi * 16);
        v16bf b = *(const v16bf*)(wr + kc * 32 + hi * 16);
        acc = __builtin_amdgcn_wmma_f32_16x16x32_bf16(false, a, false, b, (short)0, acc, false, false);
    }
    float bias = (ncol < NCLS) ? b2[ncol] : 0.0f;
#pragma unroll
    for (int i = 0; i < 8; ++i) {
        logits[(size_t)(row0 + hi * 8 + i) * CPAD + ncol] = acc[i] + bias;
    }
}

// per-node log_softmax over the first 41 (valid) columns
__global__ void k_log_softmax(const float* __restrict__ logits, float* __restrict__ out) {
    int n = blockIdx.x * blockDim.x + threadIdx.x;
    if (n >= N_NODES) return;
    const float* r = logits + (size_t)n * CPAD;
    float mx = r[0];
#pragma unroll 8
    for (int j = 1; j < NCLS; ++j) mx = fmaxf(mx, r[j]);
    float s = 0.0f;
#pragma unroll 8
    for (int j = 0; j < NCLS; ++j) s += __expf(r[j] - mx);
    float lse = mx + __logf(s);
#pragma unroll 8
    for (int j = 0; j < NCLS; ++j) out[(size_t)n * NCLS + j] = r[j] - lse;
}

// ---------- host launcher ----------

extern "C" void kernel_launch(void* const* d_in, const int* in_sizes, int n_in,
                              void* d_out, int out_size, void* d_ws, size_t ws_size,
                              hipStream_t stream) {
    const float* x   = (const float*)d_in[0];
    const float* W1l = (const float*)d_in[1];
    const float* W1r = (const float*)d_in[2];
    const float* b1  = (const float*)d_in[3];
    const float* W2l = (const float*)d_in[4];
    const float* W2r = (const float*)d_in[5];
    const float* b2  = (const float*)d_in[6];
    const int*   src = (const int*)d_in[7];
    const int*   dst = (const int*)d_in[8];
    float* out = (float*)d_out;

    // workspace carve-up (~83.2 MB)
    char* w = (char*)d_ws;
    bf16*  xb     = (bf16*)w;  w += (size_t)N_NODES * KPAD * sizeof(bf16);   // 12.8 MB
    bf16*  aggb   = (bf16*)w;  w += (size_t)N_NODES * KPAD * sizeof(bf16);   // 12.8 MB (layer1 agg bf16, reused layer2)
    bf16*  hb     = (bf16*)w;  w += (size_t)N_NODES * HID  * sizeof(bf16);   // 12.8 MB
    float* agg    = (float*)w; w += (size_t)N_NODES * KPAD * sizeof(float);  // 25.6 MB (f32 agg, reused layer2)
    float* logits = (float*)w; w += (size_t)N_NODES * CPAD * sizeof(float);  // 19.2 MB
    bf16*  Wp1l   = (bf16*)w;  w += (size_t)HID  * KPAD * sizeof(bf16);
    bf16*  Wp1r   = (bf16*)w;  w += (size_t)HID  * KPAD * sizeof(bf16);
    bf16*  Wp2l   = (bf16*)w;  w += (size_t)CPAD * KPAD * sizeof(bf16);
    bf16*  Wp2r   = (bf16*)w;  w += (size_t)CPAD * KPAD * sizeof(bf16);

    const int TB = 256;
    int packN  = 2 * HID * KPAD + 2 * CPAD * KPAD;                 // 14336
    int nodeN  = N_NODES * KPAD;                                   // 6.4M
    int edgeN  = N_EDGES * 64;                                     // 102.4M
    int rowTiles = N_NODES / 16;                                   // 6250 (exact)

    k_pack_weights<<<(packN + TB - 1) / TB, TB, 0, stream>>>(W1l, W1r, W2l, W2r,
                                                            Wp1l, Wp1r, Wp2l, Wp2r);
    k_prep<<<(nodeN + TB - 1) / TB, TB, 0, stream>>>(x, xb, agg);
    k_scatter1<<<(edgeN + TB - 1) / TB, TB, 0, stream>>>(x, src, dst, agg);
    k_cvt_zero<<<(nodeN + TB - 1) / TB, TB, 0, stream>>>(agg, aggb);
    k_gemm1<<<rowTiles, 128, 0, stream>>>(aggb, xb, Wp1l, Wp1r, b1, hb);
    k_scatter2<<<(edgeN + TB - 1) / TB, TB, 0, stream>>>(hb, src, dst, agg);
    k_cvt_zero<<<(nodeN + TB - 1) / TB, TB, 0, stream>>>(agg, aggb);   // aggb now = layer2 agg (bf16)
    k_gemm2<<<rowTiles, 96, 0, stream>>>(aggb, hb, Wp2l, Wp2r, b2, logits);
    k_log_softmax<<<(N_NODES + TB - 1) / TB, TB, 0, stream>>>(logits, out);
}
